// RLSAttn_33835752357998
// MI455X (gfx1250) — matrix operand order repaired and checked by reading, not compiled, not run on hardware
//
#include <hip/hip_runtime.h>

// ---------------------------------------------------------------------------
// CDNA5 (gfx1250) two-stage block attention, bf16 WMMA + fp32 accumulate.
// Shapes fixed by the reference: B=4,H=16,N=4096,D=64, split s=64, grid g=64.
// Workspace layout (bf16): [0,T) x1 transposed per-(b,h) as [e=s*D][f=g],
//                          [T,2T) Q bf16, [2T,3T) K bf16, T = B*H*N*D.
// ---------------------------------------------------------------------------

typedef __attribute__((ext_vector_type(16))) __bf16        v16bf;
typedef __attribute__((ext_vector_type(8)))  float         v8f;
typedef __attribute__((ext_vector_type(4)))  unsigned int  v4u;

union FragBF {
  v16bf v;
  unsigned short u[16];
  v4u q[2];
};

__device__ __forceinline__ unsigned short f2bf(float f) {
  unsigned int x = __float_as_uint(f);
  x += 0x7FFFu + ((x >> 16) & 1u);          // round-to-nearest-even
  return (unsigned short)(x >> 16);
}

__device__ __forceinline__ unsigned int pack2bf(float lo, float hi) {
  return (unsigned int)f2bf(lo) | ((unsigned int)f2bf(hi) << 16);
}

__device__ __forceinline__ v8f wmma_bf16(v16bf a, v16bf b, v8f c) {
  // v_wmma_f32_16x16x32_bf16: (neg_a, A, neg_b, B, c_mod, C, reuse_a, reuse_b)
  return __builtin_amdgcn_wmma_f32_16x16x32_bf16(false, a, false, b,
                                                 (short)0, c, false, false);
}

// ---- fragment builders (contiguous-K order, same for A and B, wave32) -------------------
// A (16x32 bf16): lane L holds row m=L&15; K = (L>>4)*16 + s
//                 -> 16 contiguous K per lane.
__device__ __forceinline__ v16bf frag_a_lds(const unsigned short* p, int row0,
                                            int kbase, int lane) {
  FragBF f;
  const unsigned short* rp =
      p + (row0 + (lane & 15)) * 64 + kbase + ((lane >> 4) << 4);
#pragma unroll
  for (int i = 0; i < 8; ++i) { f.u[i] = rp[i]; f.u[8 + i] = rp[8 + i]; }
  return f.v;
}

// B (32x16 bf16) with B[k][n] = M[n][k] (Q·K^T): 16 contiguous K per lane.
__device__ __forceinline__ v16bf frag_bt_lds(const unsigned short* p, int col0,
                                             int kbase, int lane) {
  FragBF f;
  const unsigned short* rp =
      p + (col0 + (lane & 15)) * 64 + kbase + ((lane >> 4) << 4);
#pragma unroll
  for (int i = 0; i < 16; ++i) f.u[i] = rp[i];
  return f.v;
}

// B (32x16 bf16) with B[k][n] = M[k][n] (P·V): strided LDS gathers.
__device__ __forceinline__ v16bf frag_b_lds(const unsigned short* p, int col0,
                                            int kbase, int lane) {
  FragBF f;
  const unsigned short* cp =
      p + (kbase + ((lane >> 4) << 4)) * 64 + col0 + (lane & 15);
#pragma unroll
  for (int i = 0; i < 16; ++i) f.u[i] = cp[i * 64];
  return f.v;
}

// Parallel 64x64 row softmax: 4 lanes per row, 16 elems each, registers only.
// ls: fp32 scores in, lp: bf16 probabilities out. Call with 256 threads.
__device__ __forceinline__ void softmax64(const float* ls, unsigned short* lp,
                                          int tid) {
  const int row = tid >> 2;
  const int seg = (tid & 3) << 4;
  float vals[16];
  const float* rp = ls + row * 64 + seg;
  float mx = -3.0e38f;
#pragma unroll
  for (int j = 0; j < 16; ++j) { vals[j] = rp[j]; mx = fmaxf(mx, vals[j]); }
  mx = fmaxf(mx, __shfl_xor(mx, 1));
  mx = fmaxf(mx, __shfl_xor(mx, 2));
  float sum = 0.f;
#pragma unroll
  for (int j = 0; j < 16; ++j) { vals[j] = __expf(vals[j] - mx); sum += vals[j]; }
  sum += __shfl_xor(sum, 1);
  sum += __shfl_xor(sum, 2);
  const float inv = 1.f / sum;
  unsigned short* pp = lp + row * 64 + seg;
#pragma unroll
  for (int j = 0; j < 16; ++j) pp[j] = f2bf(vals[j] * inv);
}

// ---------------------------------------------------------------------------
// Stage 1: per-(b,h,g) 64x64 SDPA. grid = B*H*g, block = 256 (8 waves).
// Also mirrors Q/K to bf16 workspace and stores x1 transposed for stage 2.
// ---------------------------------------------------------------------------
__global__ __launch_bounds__(256) void rls_stage1(
    const float* __restrict__ Q, const float* __restrict__ K,
    const float* __restrict__ V, unsigned short* __restrict__ x1t,
    unsigned short* __restrict__ qbf, unsigned short* __restrict__ kbf) {
  __shared__ unsigned short lq[4096], lk[4096], lv[4096], lp[4096];
  __shared__ float ls[4096];

  const size_t base = (size_t)blockIdx.x * 4096u;
  const int tid = threadIdx.x;

  // Load block Q/K/V (64x64 fp32), down-convert to bf16 in LDS; mirror the
  // bf16 Q/K to the workspace so stage 2 streams bf16 (no re-conversion).
  const float4* q4 = (const float4*)(Q + base);
  const float4* k4 = (const float4*)(K + base);
  const float4* v4 = (const float4*)(V + base);
  uint2* qbf2 = (uint2*)qbf + (base >> 2);
  uint2* kbf2 = (uint2*)kbf + (base >> 2);
  for (int i = tid; i < 1024; i += 256) {
    float4 q = q4[i], k = k4[i], v = v4[i];
    const int o = i << 2;
    uint2 qp, kp;
    qp.x = pack2bf(q.x, q.y); qp.y = pack2bf(q.z, q.w);
    kp.x = pack2bf(k.x, k.y); kp.y = pack2bf(k.z, k.w);
    qbf2[i] = qp; kbf2[i] = kp;
    lq[o] = (unsigned short)(qp.x); lq[o + 1] = (unsigned short)(qp.x >> 16);
    lq[o + 2] = (unsigned short)(qp.y); lq[o + 3] = (unsigned short)(qp.y >> 16);
    lk[o] = (unsigned short)(kp.x); lk[o + 1] = (unsigned short)(kp.x >> 16);
    lk[o + 2] = (unsigned short)(kp.y); lk[o + 3] = (unsigned short)(kp.y >> 16);
    lv[o] = f2bf(v.x); lv[o + 1] = f2bf(v.y); lv[o + 2] = f2bf(v.z); lv[o + 3] = f2bf(v.w);
  }
  __syncthreads();

  const int lane = tid & 31, wave = tid >> 5;
  const int mT  = wave >> 1;          // 4 m-tiles
  const int nT0 = (wave & 1) << 1;    // 2 n-tiles per wave
  const int l15 = lane & 15;
  const int h8  = (lane >> 4) << 3;

  // S = Q · K^T   (M=N=64, K=64)
  v8f s0 = {}, s1 = {};
#pragma unroll
  for (int ks = 0; ks < 2; ++ks) {
    v16bf a  = frag_a_lds(lq, mT * 16, ks * 32, lane);
    v16bf b0 = frag_bt_lds(lk, nT0 * 16, ks * 32, lane);
    v16bf b1 = frag_bt_lds(lk, nT0 * 16 + 16, ks * 32, lane);
    s0 = wmma_bf16(a, b0, s0);
    s1 = wmma_bf16(a, b1, s1);
  }
  const int mrow = mT * 16 + h8;
#pragma unroll
  for (int r = 0; r < 8; ++r) {
    ls[(mrow + r) * 64 + nT0 * 16 + l15]      = s0[r] * 0.125f;  // 1/sqrt(64)
    ls[(mrow + r) * 64 + nT0 * 16 + 16 + l15] = s1[r] * 0.125f;
  }
  __syncthreads();

  softmax64(ls, lp, tid);
  __syncthreads();

  // x1 = P · V   (M=64 s-rows, N=64 d-cols, K=64)
  v8f x0 = {}, x1 = {};
#pragma unroll
  for (int ks = 0; ks < 2; ++ks) {
    v16bf a  = frag_a_lds(lp, mT * 16, ks * 32, lane);
    v16bf b0 = frag_b_lds(lv, nT0 * 16, ks * 32, lane);
    v16bf b1 = frag_b_lds(lv, nT0 * 16 + 16, ks * 32, lane);
    x0 = wmma_bf16(a, b0, x0);
    x1 = wmma_bf16(a, b1, x1);
  }

  // Store x1 TRANSPOSED: per-(b,h) slab is [e = m*64+n][f = g], so stage 2's
  // B-fragments become contiguous 32B loads. Element (m,n) -> (m*64+n)*64 + g.
  unsigned short* xw = x1t + (size_t)(blockIdx.x >> 6) * 262144u
                           + (size_t)(blockIdx.x & 63);
#pragma unroll
  for (int r = 0; r < 8; ++r) {
    xw[(size_t)((mrow + r) * 64 + nT0 * 16 + l15) * 64u]      = f2bf(x0[r]);
    xw[(size_t)((mrow + r) * 64 + nT0 * 16 + 16 + l15) * 64u] = f2bf(x1[r]);
  }
}

// ---------------------------------------------------------------------------
// Stage 2: per-(b,h) block-summary attention. grid = B*H, block = 256.
//   scores2 = q2 · k2^T / 64  (64x64, K-dim 4096) from bf16 mirrors,
//   softmax, x = attn2 · x1T  (64x4096, K-dim 64).
// ---------------------------------------------------------------------------
__global__ __launch_bounds__(256) void rls_stage2(
    const unsigned short* __restrict__ qbf, const unsigned short* __restrict__ kbf,
    const unsigned short* __restrict__ x1t, float* __restrict__ out) {
  __shared__ float ls[4096];
  __shared__ unsigned short lp[4096];

  const size_t base = (size_t)blockIdx.x * 262144u;  // 64 * 4096
  const int tid = threadIdx.x, lane = tid & 31, wave = tid >> 5;
  const int mT  = wave >> 1;
  const int nT0 = (wave & 1) << 1;
  const int l15 = lane & 15;
  const int h8  = (lane >> 4) << 3;
  const int h16 = (lane >> 4) << 4;

  // --- scores2: 64x64 output, 128 K-steps of 32; pure b128 loads + WMMA.
  const unsigned short* qrow  = qbf + base + (size_t)(mT * 16 + l15) * 4096;
  const unsigned short* krow0 = kbf + base + (size_t)(nT0 * 16 + l15) * 4096;
  const unsigned short* krow1 = krow0 + (size_t)16 * 4096;
  v8f s0 = {}, s1 = {};
#pragma unroll 2
  for (int ks = 0; ks < 128; ++ks) {
    const int kb = ks * 32;
    __builtin_prefetch(qrow + kb + 512, 0, 0);   // global_prefetch_b8
    __builtin_prefetch(krow0 + kb + 512, 0, 0);
    __builtin_prefetch(krow1 + kb + 512, 0, 0);
    FragBF a, b0, b1;
    const v4u* ap  = (const v4u*)(qrow + kb + h16);  // 16 contiguous elems
    a.q[0] = ap[0]; a.q[1] = ap[1];
    const v4u* bp0 = (const v4u*)(krow0 + kb + h16); // 16 contiguous elems
    b0.q[0] = bp0[0]; b0.q[1] = bp0[1];
    const v4u* bp1 = (const v4u*)(krow1 + kb + h16);
    b1.q[0] = bp1[0]; b1.q[1] = bp1[1];
    s0 = wmma_bf16(a.v, b0.v, s0);
    s1 = wmma_bf16(a.v, b1.v, s1);
  }
  const int mrow = mT * 16 + h8;
#pragma unroll
  for (int r = 0; r < 8; ++r) {
    ls[(mrow + r) * 64 + nT0 * 16 + l15]      = s0[r] * 0.015625f;  // 1/sqrt(4096)
    ls[(mrow + r) * 64 + nT0 * 16 + 16 + l15] = s1[r] * 0.015625f;
  }
  __syncthreads();

  softmax64(ls, lp, tid);
  __syncthreads();

  // Hoist all 8 attn2 A-fragments (4 m-tiles x 2 K-steps) into registers.
  v16bf afr[4][2];
#pragma unroll
  for (int mt = 0; mt < 4; ++mt)
#pragma unroll
    for (int ks = 0; ks < 2; ++ks)
      afr[mt][ks] = frag_a_lds(lp, mt * 16, ks * 32, lane);

  // x = attn2 · x1_grid: each wave owns 32 of the 256 n-tiles over the 4096
  // summary dim. x1T rows give contiguous 32B B-fragment loads.
  const unsigned short* xb = x1t + base;
  for (int nn = wave * 32; nn < wave * 32 + 32; ++nn) {
    const int e = nn * 16 + l15;
    const v4u* xrow = (const v4u*)(xb + (size_t)e * 64u);  // 64 bf16 per row
    FragBF b[2];
#pragma unroll
    for (int ks = 0; ks < 2; ++ks) {
      const int off = (ks * 32 + h16) >> 3;  // in v4u units (8 bf16 each)
      b[ks].q[0] = xrow[off]; b[ks].q[1] = xrow[off + 1];
    }
    v8f acc[4] = {{}, {}, {}, {}};
#pragma unroll
    for (int ks = 0; ks < 2; ++ks)
#pragma unroll
      for (int mt = 0; mt < 4; ++mt)
        acc[mt] = wmma_bf16(afr[mt][ks], b[ks].v, acc[mt]);
#pragma unroll
    for (int mt = 0; mt < 4; ++mt) {
      float* op = out + base + (size_t)(mt * 16 + h8) * 4096 + e;
#pragma unroll
      for (int r = 0; r < 8; ++r) op[(size_t)r * 4096] = acc[mt][r];
    }
  }
}

// ---------------------------------------------------------------------------
extern "C" void kernel_launch(void* const* d_in, const int* in_sizes, int n_in,
                              void* d_out, int out_size, void* d_ws,
                              size_t ws_size, hipStream_t stream) {
  (void)n_in; (void)out_size; (void)ws_size;
  const float* Q = (const float*)d_in[0];
  const float* K = (const float*)d_in[1];
  const float* V = (const float*)d_in[2];
  float* out = (float*)d_out;

  const int total   = in_sizes[0];       // B*H*N*D = 16,777,216
  const int nblocks = total / 4096;      // B*H*g   = 4096
  const int nbh     = total / 262144;    // B*H     = 64

  unsigned short* x1t = (unsigned short*)d_ws;       // bf16, total elems
  unsigned short* qbf = x1t + total;                 // bf16 Q mirror
  unsigned short* kbf = qbf + total;                 // bf16 K mirror

  rls_stage1<<<nblocks, 256, 0, stream>>>(Q, K, V, x1t, qbf, kbf);
  rls_stage2<<<nbh, 256, 0, stream>>>(qbf, kbf, x1t, out);
}
